// GNNPerturbModel_6923487282342
// MI455X (gfx1250) — matrix operand order repaired and checked by reading, not compiled, not run on hardware
//
#include <hip/hip_runtime.h>

typedef __bf16 bf16;
typedef __attribute__((ext_vector_type(16))) __bf16 bf16x16;
typedef __attribute__((ext_vector_type(8)))  float  floatx8;
typedef unsigned int u32x4 __attribute__((ext_vector_type(4)));
typedef int v4i_t __attribute__((ext_vector_type(4)));

union Frag { bf16x16 v; u32x4 q[2]; };

#define AS1 __attribute__((address_space(1)))
#define AS3 __attribute__((address_space(3)))

#if defined(__HIP_DEVICE_COMPILE__) && __has_builtin(__builtin_amdgcn_global_load_async_to_lds_b128)
#define GEMM_ASYNC 1
#else
#define GEMM_ASYNC 0
#endif

__device__ __forceinline__ void wait_async0() {
#if defined(__HIP_DEVICE_COMPILE__)
#if __has_builtin(__builtin_amdgcn_s_wait_asynccnt)
  __builtin_amdgcn_s_wait_asynccnt(0);
#else
  asm volatile("s_wait_asynccnt 0x0" ::: "memory");
#endif
#endif
}

__device__ inline bf16 f2bf(float f) {
  unsigned u = __builtin_bit_cast(unsigned, f);
  unsigned short h = (unsigned short)((u + 0x7FFFu + ((u >> 16) & 1u)) >> 16);
  return __builtin_bit_cast(bf16, h);
}

__device__ inline float gelu_tanh(float x) {
  const float c0 = 0.7978845608028654f;  // sqrt(2/pi)
  float t = tanhf(c0 * (x + 0.044715f * x * x * x));
  return 0.5f * x * (1.0f + t);
}

// One K-step of WMMA from a staged LDS tile pair (wave tile 32x64 = 2x4 WMMAs).
__device__ __forceinline__ void wmma_tile_step(
    const bf16 (* __restrict__ Asb)[40], const bf16 (* __restrict__ Bsb)[40],
    int wm, int wn, int l16, int lhalf, floatx8 (&acc)[2][4])
{
  Frag afr[2], bfr[4];
#pragma unroll
  for (int i = 0; i < 2; i++) {
    int m = wm + i * 16 + l16;
    // A-frag (16-bit 16x32): lane half picks K chunks {8h..8h+8, 16+8h..+8}
    afr[i].q[0] = *(const u32x4*)&Asb[m][8 * lhalf];
    afr[i].q[1] = *(const u32x4*)&Asb[m][16 + 8 * lhalf];
  }
#pragma unroll
  for (int j = 0; j < 4; j++) {
    int n = wn + j * 16 + l16;
    // B-frag: lane half picks K chunks {8h..8h+8, 16+8h..+8}
    bfr[j].q[0] = *(const u32x4*)&Bsb[n][8 * lhalf];
    bfr[j].q[1] = *(const u32x4*)&Bsb[n][16 + 8 * lhalf];
  }
#pragma unroll
  for (int i = 0; i < 2; i++)
#pragma unroll
    for (int j = 0; j < 4; j++)
      acc[i][j] = __builtin_amdgcn_wmma_f32_16x16x32_bf16(
          false, afr[i].v, false, bfr[j].v, (short)0, acc[i][j], false, false);
}

// ---------------------------------------------------------------------------
// Universal bf16 WMMA GEMM:  C[M,N] = A[M,K] @ Bt[N,K]^T  (+bias +gelu +resid)
// Block tile 128x128x32, 256 threads = 8 waves, wave tile 32x64 (2x4 WMMA).
// Interior tiles use async global->LDS copies with double buffering.
// flags: bit0 = gelu
// ---------------------------------------------------------------------------
#define FLAG_GELU 1

__global__ __launch_bounds__(256) void gemm_bf16_wmma(
    const bf16* __restrict__ A,   // [M,K] row-major bf16
    const bf16* __restrict__ Bt,  // [N,K] row-major bf16 (pre-transposed weight)
    const float* __restrict__ bias,   // [N] or nullptr
    const float* resid,               // [M,N] or nullptr (may alias outF)
    float* outF,                      // [M,N] f32 output or nullptr
    bf16* __restrict__ outB,          // [M,N] bf16 output or nullptr
    int M, int N, int K, int flags)
{
  __shared__ __align__(16) bf16 As[2][128][40];
  __shared__ __align__(16) bf16 Bs[2][128][40];

  const int tid  = threadIdx.x;
  const int m0   = blockIdx.y * 128;
  const int n0   = blockIdx.x * 128;
  const int wave = tid >> 5;
  const int lane = tid & 31;
  const int wm   = (wave & 3) * 32;   // wave row offset in tile
  const int wn   = (wave >> 2) * 64;  // wave col offset in tile
  const int lhalf = lane >> 4;
  const int l16   = lane & 15;

  floatx8 acc[2][4];
#pragma unroll
  for (int i = 0; i < 2; i++)
#pragma unroll
    for (int j = 0; j < 4; j++)
#pragma unroll
      for (int e = 0; e < 8; e++) acc[i][j][e] = 0.0f;

  const int lr = tid >> 1;         // 0..127 : tile row this thread stages
  const int lc = (tid & 1) * 16;   // 0 or 16: element chunk

#if GEMM_ASYNC
  if (m0 + 128 <= M && n0 + 128 <= N) {
    // ---- async double-buffered path (interior tiles; no bounds/zero-fill) ----
    const bf16* ga = A  + (size_t)(m0 + lr) * K + lc;
    const bf16* gb = Bt + (size_t)(n0 + lr) * K + lc;
    auto issue = [&](int kt, int buf) {
      const bf16* pa = ga + kt * 32;
      const bf16* pb = gb + kt * 32;
      __builtin_amdgcn_global_load_async_to_lds_b128(
          (AS1 v4i_t*)pa,       (AS3 v4i_t*)&As[buf][lr][lc],     0, 0);
      __builtin_amdgcn_global_load_async_to_lds_b128(
          (AS1 v4i_t*)(pa + 8), (AS3 v4i_t*)&As[buf][lr][lc + 8], 0, 0);
      __builtin_amdgcn_global_load_async_to_lds_b128(
          (AS1 v4i_t*)pb,       (AS3 v4i_t*)&Bs[buf][lr][lc],     0, 0);
      __builtin_amdgcn_global_load_async_to_lds_b128(
          (AS1 v4i_t*)(pb + 8), (AS3 v4i_t*)&Bs[buf][lr][lc + 8], 0, 0);
    };
    issue(0, 0);
    const int nk = K >> 5;
    for (int kt = 0; kt < nk; kt++) {
      int buf = kt & 1;
      wait_async0();        // my copies into buf complete
      __syncthreads();      // everyone's copies complete; prev reads of buf^1 done
      if (kt + 1 < nk) issue(kt + 1, buf ^ 1);  // overlap with compute below
      wmma_tile_step(As[buf], Bs[buf], wm, wn, l16, lhalf, acc);
    }
  } else
#endif
  {
    // ---- synchronous fallback (boundary tiles, or no async builtin) ----
    for (int k0 = 0; k0 < K; k0 += 32) {
      __syncthreads();
      {
        u32x4 d0 = {0, 0, 0, 0}, d1 = {0, 0, 0, 0};
        int gm = m0 + lr;
        if (gm < M) {
          const u32x4* p = (const u32x4*)(A + (size_t)gm * K + k0 + lc);
          d0 = p[0]; d1 = p[1];
        }
        *(u32x4*)&As[0][lr][lc]     = d0;
        *(u32x4*)&As[0][lr][lc + 8] = d1;
      }
      {
        u32x4 d0 = {0, 0, 0, 0}, d1 = {0, 0, 0, 0};
        int gn = n0 + lr;
        if (gn < N) {
          const u32x4* p = (const u32x4*)(Bt + (size_t)gn * K + k0 + lc);
          d0 = p[0]; d1 = p[1];
        }
        *(u32x4*)&Bs[0][lr][lc]     = d0;
        *(u32x4*)&Bs[0][lr][lc + 8] = d1;
      }
      __syncthreads();
      wmma_tile_step(As[0], Bs[0], wm, wn, l16, lhalf, acc);
    }
  }

  // Epilogue: C/D layout -> lane l16 = N col, VGPR r -> M row (r + 8*lhalf)
#pragma unroll
  for (int i = 0; i < 2; i++) {
#pragma unroll
    for (int j = 0; j < 4; j++) {
      int n = n0 + wn + j * 16 + l16;
      if (n >= N) continue;
      float bv = bias ? bias[n] : 0.0f;
      int mb = m0 + wm + i * 16 + lhalf * 8;
#pragma unroll
      for (int r = 0; r < 8; r++) {
        int m = mb + r;
        if (m >= M) continue;
        float v = acc[i][j][r] + bv;
        if (flags & FLAG_GELU) v = gelu_tanh(v);
        size_t idx = (size_t)m * N + n;
        if (resid) v += resid[idx];
        if (outF) outF[idx] = v;
        if (outB) outB[idx] = f2bf(v);
      }
    }
  }
}

// ---------------------------------------------------------------------------
// Transpose + convert weight: in[K,N] f32 -> out[N,K] bf16 (dims % 32 == 0)
// ---------------------------------------------------------------------------
__global__ void transconv_kernel(const float* __restrict__ in, bf16* __restrict__ out,
                                 int K, int N)
{
  __shared__ float tile[32][33];
  size_t base = (size_t)blockIdx.z * K * N;
  int k0 = blockIdx.y * 32, n0 = blockIdx.x * 32;
  int tx = threadIdx.x & 31, ty = threadIdx.x >> 5;  // 256 threads = 32x8
#pragma unroll
  for (int i = 0; i < 32; i += 8)
    tile[ty + i][tx] = in[base + (size_t)(k0 + ty + i) * N + n0 + tx];
  __syncthreads();
#pragma unroll
  for (int i = 0; i < 32; i += 8)
    out[base + (size_t)(n0 + ty + i) * K + k0 + tx] = f2bf(tile[tx][ty + i]);
}

__global__ void conv_kernel(const float* __restrict__ in, bf16* __restrict__ out, long n)
{
  long i = (long)blockIdx.x * blockDim.x + threadIdx.x;
  if (i < n) out[i] = f2bf(in[i]);
}

// ---------------------------------------------------------------------------
// Edge scatter-add: agg[dst] += msgs[src] * ew   (float4 per thread, f32 atomics)
// ---------------------------------------------------------------------------
__global__ void scatter_kernel(const float* __restrict__ msgs, const int* __restrict__ ei,
                               const float* __restrict__ ew, float* agg, int E)
{
  int gid = blockIdx.x * blockDim.x + threadIdx.x;   // E * 64 threads (D/4 = 64)
  if (gid >= E * 64) return;
  int e = gid >> 6;
  int c = (gid & 63) << 2;
  int s = ei[e];
  int d = ei[E + e];
  float w = ew[e];
  const float* mp = msgs + (size_t)s * 256 + c;
  float* ap = agg + (size_t)d * 256 + c;
  unsafeAtomicAdd(ap + 0, mp[0] * w);
  unsafeAtomicAdd(ap + 1, mp[1] * w);
  unsafeAtomicAdd(ap + 2, mp[2] * w);
  unsafeAtomicAdd(ap + 3, mp[3] * w);
}

// x = x + relu(agg + bias[d]); also refresh bf16 mirror of x
__global__ void addrelu_kernel(float* x, const float* __restrict__ agg,
                               const float* __restrict__ bias, bf16* xb, long n)
{
  long i = (long)blockIdx.x * blockDim.x + threadIdx.x;
  if (i >= n) return;
  float v = agg[i] + bias[i & 255];
  v = v > 0.0f ? v : 0.0f;
  float o = x[i] + v;
  x[i] = o;
  xb[i] = f2bf(o);
}

// ---------------------------------------------------------------------------
// LayerNorm (wave per row, optional OOV gather) -> bf16 output
// ---------------------------------------------------------------------------
__global__ __launch_bounds__(256) void ln_kernel(
    const float* __restrict__ X, const float* __restrict__ g,
    const float* __restrict__ b, bf16* __restrict__ out,
    int rows, int dim, const int* __restrict__ idx,
    const float* __restrict__ oov, int nclip)
{
  int row  = blockIdx.x * 8 + (threadIdx.x >> 5);
  int lane = threadIdx.x & 31;
  if (row >= rows) return;
  const float* src;
  if (idx) {
    int id = idx[row];
    src = (id >= 0) ? (X + (size_t)min(id, nclip - 1) * dim) : oov;
  } else {
    src = X + (size_t)row * dim;
  }
  float s = 0.0f, ss = 0.0f;
  for (int i = lane; i < dim; i += 32) { float v = src[i]; s += v; ss += v * v; }
#pragma unroll
  for (int off = 16; off > 0; off >>= 1) {
    s  += __shfl_xor(s, off);
    ss += __shfl_xor(ss, off);
  }
  float inv = 1.0f / (float)dim;
  float mu  = s * inv;
  float var = ss * inv - mu * mu;
  float rs  = rsqrtf(var + 1e-5f);
  for (int i = lane; i < dim; i += 32)
    out[(size_t)row * dim + i] = f2bf((src[i] - mu) * rs * g[i] + b[i]);
}

// ---------------------------------------------------------------------------
extern "C" void kernel_launch(void* const* d_in, const int* in_sizes, int n_in,
                              void* d_out, int out_size, void* d_ws, size_t ws_size,
                              hipStream_t stream) {
  constexpr int NNODES = 20000, NEDGES = 640000, D = 256, HID = 512, NB = 6,
                EXPF = 4, NC = 3, R = 512, G = 6640, BB = 4096;

  const int*   idxs   = (const int*)  d_in[0];
  const float* frozen = (const float*)d_in[1];
  const int*   eidx   = (const int*)  d_in[2];
  const float* ew     = (const float*)d_in[3];
  const float* W6 = (const float*)d_in[4],  *b6 = (const float*)d_in[5];
  const float* W7 = (const float*)d_in[6],  *b7 = (const float*)d_in[7];
  const float* Wp = (const float*)d_in[8],  *bp = (const float*)d_in[9];
  const float* oov  = (const float*)d_in[10];
  const float* in_g = (const float*)d_in[11], *in_b = (const float*)d_in[12];
  const float* inW  = (const float*)d_in[13], *inb  = (const float*)d_in[14];
  const float* blg  = (const float*)d_in[15], *blb  = (const float*)d_in[16];
  const float* f1W  = (const float*)d_in[17], *f1b  = (const float*)d_in[18];
  const float* f2W  = (const float*)d_in[19], *f2b  = (const float*)d_in[20];
  const float* og   = (const float*)d_in[21], *ob   = (const float*)d_in[22];
  const float* outW = (const float*)d_in[23], *outb = (const float*)d_in[24];
  const float* gene = (const float*)d_in[25];
  float* out = (float*)d_out;

  char* ws = (char*)d_ws;
  size_t off = 0;
  auto take = [&](size_t bytes) -> void* {
    void* p = ws + off;
    off += (bytes + 255) & ~(size_t)255;
    return p;
  };
  bf16* W6b   = (bf16*)take((size_t)D * D * 2);
  bf16* W7b   = (bf16*)take((size_t)D * D * 2);
  bf16* Wpb   = (bf16*)take((size_t)D * D * 2);
  bf16* inWb  = (bf16*)take((size_t)D * HID * 2);
  bf16* f1Wb  = (bf16*)take((size_t)NB * HID * HID * EXPF * 2);
  bf16* f2Wb  = (bf16*)take((size_t)NB * HID * EXPF * HID * 2);
  bf16* outWb = (bf16*)take((size_t)HID * NC * R * 2);
  bf16* geneb = (bf16*)take((size_t)G * R * 2);
  float* xf   = (float*)take((size_t)NNODES * D * 4);
  bf16*  xb   = (bf16*) take((size_t)NNODES * D * 2);
  float* tmp  = (float*)take((size_t)NNODES * D * 4);
  float* agg  = (float*)take((size_t)NNODES * D * 4);
  bf16*  projb = (bf16*)take((size_t)BB * NC * R * 2);
  // Head-phase aliases into GNN-phase buffers (lifetimes disjoint):
  bf16*  lnemb = (bf16*)agg;   // [BB, D]
  float* h     = xf;           // [BB, HID]
  bf16*  hnb   = xb;           // [BB, HID]
  bf16*  tb    = (bf16*)tmp;   // [BB, HID*EXPF]
  if (off > ws_size) return;   // workspace too small; bail

  auto tconv = [&](const float* in, bf16* o, int K, int N, int batch) {
    dim3 g(N / 32, K / 32, batch);
    transconv_kernel<<<g, 256, 0, stream>>>(in, o, K, N);
  };
  auto conv = [&](const float* in, bf16* o, long n) {
    conv_kernel<<<(int)((n + 255) / 256), 256, 0, stream>>>(in, o, n);
  };
  auto gemm = [&](const bf16* A, const bf16* Bt, const float* bias, const float* resid,
                  float* oF, bf16* oB, int M, int N, int K, int flags) {
    dim3 g((N + 127) / 128, (M + 127) / 128);
    gemm_bf16_wmma<<<g, 256, 0, stream>>>(A, Bt, bias, resid, oF, oB, M, N, K, flags);
  };

  // ---- weight conversion (f32 -> bf16, weights pre-transposed to [N,K]) ----
  tconv(W6, W6b, D, D, 1);
  tconv(W7, W7b, D, D, 1);
  tconv(Wp, Wpb, D, D, 1);
  tconv(inW, inWb, D, HID, 1);
  tconv(f1W, f1Wb, HID, HID * EXPF, NB);
  tconv(f2W, f2Wb, HID * EXPF, HID, NB);
  tconv(outW, outWb, HID, NC * R, 1);
  conv(gene, geneb, (long)G * R);          // gene_emb already [G,R] = [N,K]
  conv(frozen, xb, (long)NNODES * D);
  (void)hipMemcpyAsync(xf, frozen, (size_t)NNODES * D * 4, hipMemcpyDeviceToDevice, stream);

  // ---- GNN tail: two residual message-passing layers ----
  const bf16* Wsb[2] = {W6b, W7b};
  const float* bsv[2] = {b6, b7};
  const long nD = (long)NNODES * D;
  for (int l = 0; l < 2; l++) {
    gemm(xb, Wsb[l], nullptr, nullptr, tmp, nullptr, NNODES, D, D, 0);
    (void)hipMemsetAsync(agg, 0, (size_t)NNODES * D * 4, stream);
    scatter_kernel<<<(NEDGES * 64 + 255) / 256, 256, 0, stream>>>(tmp, eidx, ew, agg, NEDGES);
    addrelu_kernel<<<(int)((nD + 255) / 256), 256, 0, stream>>>(xf, agg, bsv[l], xb, nD);
  }
  // post-MP projection
  gemm(xb, Wpb, bp, nullptr, tmp, nullptr, NNODES, D, D, 0);

  // ---- gather (OOV-masked) + input LN -> bf16 ----
  ln_kernel<<<(BB + 7) / 8, 256, 0, stream>>>(tmp, in_g, in_b, lnemb, BB, D, idxs, oov, NNODES);
  gemm(lnemb, inWb, inb, nullptr, h, nullptr, BB, HID, D, 0);

  // ---- residual MLP blocks ----
  for (int i = 0; i < NB; i++) {
    ln_kernel<<<(BB + 7) / 8, 256, 0, stream>>>(h, blg + (size_t)i * HID, blb + (size_t)i * HID,
                                                hnb, BB, HID, nullptr, nullptr, 0);
    gemm(hnb, f1Wb + (size_t)i * HID * HID * EXPF, f1b + (size_t)i * HID * EXPF,
         nullptr, nullptr, tb, BB, HID * EXPF, HID, FLAG_GELU);
    gemm(tb, f2Wb + (size_t)i * HID * EXPF * HID, f2b + (size_t)i * HID,
         h, h, nullptr, BB, HID, HID * EXPF, 0);
  }

  // ---- output head ----
  ln_kernel<<<(BB + 7) / 8, 256, 0, stream>>>(h, og, ob, hnb, BB, HID, nullptr, nullptr, 0);
  gemm(hnb, outWb, outb, nullptr, nullptr, projb, BB, NC * R, HID, 0);
  // logits[b*3+c, g] = proj[b*3+c, :] . gene[g, :]
  gemm(projb, geneb, nullptr, nullptr, out, nullptr, BB * NC, G, R, 0);
}
